// CLAM_SB_75436805587054
// MI455X (gfx1250) — matrix operand order repaired and checked by reading, not compiled, not run on hardware
//
#include <hip/hip_runtime.h>
#include <hip/hip_bf16.h>

typedef __attribute__((ext_vector_type(16))) __bf16 v16bf;
typedef __attribute__((ext_vector_type(8)))  __bf16 v8bf;
typedef __attribute__((ext_vector_type(8)))  float  v8f;
typedef __attribute__((ext_vector_type(4)))  unsigned int u32x4;
typedef __attribute__((ext_vector_type(8)))  int i32x8;
typedef __attribute__((ext_vector_type(4)))  int i32x4;

#define NROWS 50000
#define MTILES 3125           // 50000 / 16

#if defined(__AMDGCN__) && __has_builtin(__builtin_amdgcn_tensor_load_to_lds) && \
    __has_builtin(__builtin_amdgcn_s_wait_tensorcnt)
#define CLAM_USE_TDM 1
#else
#define CLAM_USE_TDM 0
#endif

__device__ __forceinline__ unsigned short f2bf(float f) {
  unsigned int u = __float_as_uint(f);
  unsigned int r = u + 0x7FFFu + ((u >> 16) & 1u);   // round-to-nearest-even
  return (unsigned short)(r >> 16);
}
__device__ __forceinline__ float bf2f(unsigned short u) {
  return __uint_as_float(((unsigned int)u) << 16);
}
__device__ __forceinline__ v8f wmma_bf16(v16bf a, v16bf b, v8f c) {
  return __builtin_amdgcn_wmma_f32_16x16x32_bf16(false, a, false, b, (short)0, c,
                                                 false, false);
}
// Build A fragment (16x32 bf16) from row-major bf16 memory with given row stride.
// Layout: lane<16 -> M=lane, K = {0..15}; lane>=16 -> M=lane-16, K = {16..31}
__device__ __forceinline__ v16bf load_a_frag(const unsigned short* base, int stride,
                                             int lane, int kb) {
  int m    = lane & 15;
  int koff = (lane >> 4) << 4;                       // 0 or 16
  const unsigned short* rp = base + m * stride + kb;
  v8bf lo = *(const v8bf*)(rp + koff);
  v8bf hi = *(const v8bf*)(rp + 8 + koff);
  return __builtin_shufflevector(lo, hi, 0,1,2,3,4,5,6,7,8,9,10,11,12,13,14,15);
}

// ---------------------------------------------------------------------------
// Pack f32 weight [K][Nc] row-major into bf16 B-fragment tiles (32x16 per tile).
// Tile (kt,nt): element index = lane*16 + i, where n = nt*16 + lane%16,
// k = kt*32 + 16*(lane/16) + i   (contiguous k order, same as load_a_frag).
// ---------------------------------------------------------------------------
__global__ __launch_bounds__(256) void clam_pack_b(const float* __restrict__ W,
                                                   unsigned short* __restrict__ out,
                                                   int K, int Nc) {
  int idx = blockIdx.x * 256 + threadIdx.x;
  if (idx >= K * Nc) return;
  int tile = idx >> 9;         // 512 elems per tile
  int r    = idx & 511;
  int l    = r >> 4;           // lane 0..31
  int i    = r & 15;           // element within lane
  int ntn  = Nc >> 4;
  int kt   = tile / ntn, nt = tile % ntn;
  int k    = kt * 32 + ((l >> 4) << 4) + i;
  int n    = nt * 16 + (l & 15);
  out[idx] = f2bf(W[(size_t)k * Nc + n]);
}

// ---------------------------------------------------------------------------
// Kernel 1: feat = relu(h @ W1 + b1), bf16 WMMA, feat stored as bf16 [N][512].
// One block = 16 rows; 8 waves x 4 N-tiles = 512 cols; K = 1024.
// ---------------------------------------------------------------------------
__global__ __launch_bounds__(256) void clam_k1_feat(const float* __restrict__ h,
                                                    const unsigned short* __restrict__ W1p,
                                                    const float* __restrict__ b1,
                                                    unsigned short* __restrict__ feat) {
  __shared__ unsigned short sh[16 * 1024];           // 32 KB bf16 h tile
  int t  = threadIdx.x;
  int mt = blockIdx.x;
  const float* hrow = h + (size_t)mt * 16 * 1024;
  for (int e = t; e < 4096; e += 256) {              // 16384 f32 as float4, cvt->bf16
    float4 v = ((const float4*)hrow)[e];
    int b = e * 4;
    sh[b+0] = f2bf(v.x); sh[b+1] = f2bf(v.y); sh[b+2] = f2bf(v.z); sh[b+3] = f2bf(v.w);
  }
  __syncthreads();

  int wave = t >> 5, lane = t & 31;
  int nt0  = wave * 4;
  v8f acc[4] = {};
  for (int kb = 0; kb < 1024; kb += 32) {
    v16bf a = load_a_frag(sh, 1024, lane, kb);
    int kt = kb >> 5;
#pragma unroll
    for (int q = 0; q < 4; q++) {
      v16bf b = *(const v16bf*)(W1p + (size_t)(kt * 32 + nt0 + q) * 512 + lane * 16);
      acc[q] = wmma_bf16(a, b, acc[q]);
    }
  }
  int colLane = lane & 15;
  int rowOff  = (lane >> 4) * 8;
#pragma unroll
  for (int q = 0; q < 4; q++) {
    int n = (nt0 + q) * 16 + colLane;
    float bias = b1[n];
#pragma unroll
    for (int v = 0; v < 8; v++) {
      float x = acc[q][v] + bias;
      x = x > 0.f ? x : 0.f;
      int row = mt * 16 + rowOff + v;
      feat[(size_t)row * 512 + n] = f2bf(x);
    }
  }
}

// ---------------------------------------------------------------------------
// Kernel 2: per 16-row tile: a=tanh(feat@Wa+ba), g=sigmoid(feat@Wb+bb),
// score=(a*g)@Wc+bc, flog=feat@Wcls+bcls. Emits scores[i], sig*flog0, sig*flog1.
// Feat tile (16 KB contiguous bf16) staged via Tensor Data Mover when available.
// ---------------------------------------------------------------------------
__global__ __launch_bounds__(256) void clam_k2_attn(const unsigned short* __restrict__ feat,
                                                    const unsigned short* __restrict__ Wap,
                                                    const unsigned short* __restrict__ Wbp,
                                                    const float* __restrict__ ba,
                                                    const float* __restrict__ bb,
                                                    const float* __restrict__ Wc,
                                                    const float* __restrict__ bc,
                                                    const float* __restrict__ Wcls,
                                                    const float* __restrict__ bcls,
                                                    float* __restrict__ scores,
                                                    float* __restrict__ f0,
                                                    float* __restrict__ f1) {
  __shared__ unsigned short sf[16 * 512];            // 16 KB feat tile (bf16)
  __shared__ float ps[16 * 256];                     // gated contributions
  __shared__ float pr[256];                          // row-sum partials
  __shared__ float pf[256];                          // Wcls dot partials
  __shared__ float fl[16 * 2];
  __shared__ float rs[16];
  int t  = threadIdx.x;
  int mt = blockIdx.x;
  const unsigned short* fr = feat + (size_t)mt * 16 * 512;

#if CLAM_USE_TDM
  if (t < 32) {
    // Tensor DMA descriptor (ISA 8.3/8.4): 1-D tile of 8192 bf16 elements.
    unsigned long long ga = (unsigned long long)(uintptr_t)fr;
    unsigned int lds = (unsigned int)(uintptr_t)&sf[0];   // LDS aperture: low 32b = offset
    u32x4 g0;
    g0[0] = 1u;                                           // count=1, user mode
    g0[1] = lds;                                          // lds_addr
    g0[2] = (unsigned int)ga;                             // global_addr[95:64]
    g0[3] = (unsigned int)((ga >> 32) & 0x01FFFFFFu) | (2u << 30);  // addr msbs | type=2
    i32x8 g1;
    g1[0] = (int)(1u << 16);                              // data_size = 2 bytes
    g1[1] = (int)(0x2000u << 16);                         // tensor_dim0 = 8192 (low 16)
    g1[2] = (int)(1u << 16);                              // tensor_dim1 = 1
    g1[3] = (int)(0x2000u << 16);                         // tile_dim0 = 8192
    g1[4] = 1;                                            // tile_dim1 = 1
    g1[5] = 8192;                                         // tensor_dim0_stride
    g1[6] = 0; g1[7] = 0;
    i32x4 z4 = {0, 0, 0, 0};
    i32x8 z8 = {0, 0, 0, 0, 0, 0, 0, 0};
    __builtin_amdgcn_tensor_load_to_lds(g0, g1, z4, z4, z8, 0);   // 6-arg form
    __builtin_amdgcn_s_wait_tensorcnt(0);
  }
#else
  for (int e = t; e < 2048; e += 256) ((uint2*)sf)[e] = ((const uint2*)fr)[e];
#endif
  __syncthreads();

  int wave = t >> 5, lane = t & 31;
  int nt0  = wave * 2;
  v8f aa[2] = {}, ag[2] = {};
  for (int kb = 0; kb < 512; kb += 32) {
    v16bf a = load_a_frag(sf, 512, lane, kb);
    int kt = kb >> 5;
#pragma unroll
    for (int q = 0; q < 2; q++) {
      size_t tile = (size_t)(kt * 16 + nt0 + q) * 512 + lane * 16;
      v16bf bA = *(const v16bf*)(Wap + tile);
      v16bf bB = *(const v16bf*)(Wbp + tile);
      aa[q] = wmma_bf16(a, bA, aa[q]);
      ag[q] = wmma_bf16(a, bB, ag[q]);
    }
  }
  int colLane = lane & 15;
  int rowOff  = (lane >> 4) * 8;
#pragma unroll
  for (int q = 0; q < 2; q++) {
    int n = (nt0 + q) * 16 + colLane;
    float bav = ba[n], bbv = bb[n], wcv = Wc[n];
#pragma unroll
    for (int v = 0; v < 8; v++) {
      float av = tanhf(aa[q][v] + bav);
      float gv = 1.f / (1.f + expf(-(ag[q][v] + bbv)));
      ps[(rowOff + v) * 256 + n] = av * gv * wcv;
    }
  }
  __syncthreads();
  {                                                  // row sums: 16-elem partials
    int r = t >> 4, seg = t & 15;
    float s = 0.f;
    for (int c = seg * 16; c < seg * 16 + 16; c++) s += ps[r * 256 + c];
    pr[t] = s;
  }
  {                                                  // feat @ Wcls: 64-elem partials
    int r = t >> 4, c = (t >> 3) & 1, seg = t & 7;
    float s = 0.f;
    for (int k = seg * 64; k < seg * 64 + 64; k++)
      s += bf2f(sf[r * 512 + k]) * Wcls[k * 2 + c];
    pf[t] = s;
  }
  __syncthreads();
  if (t < 16) {                                      // fixed-order combine
    float s = 0.f;
    for (int seg = 0; seg < 16; seg++) s += pr[t * 16 + seg];
    rs[t] = s;
  }
  if (t < 32) {
    int r = t >> 1, c = t & 1;
    float s = 0.f;
    for (int seg = 0; seg < 8; seg++) s += pf[r * 16 + c * 8 + seg];
    fl[r * 2 + c] = s + bcls[c];
  }
  __syncthreads();
  if (t < 16) {
    int g = mt * 16 + t;
    float score = rs[t] + bc[0];
    scores[g] = score;
    float sig = 1.f / (1.f + expf(-score));
    f0[g] = sig * fl[t * 2 + 0];
    f1[g] = sig * fl[t * 2 + 1];
  }
}

// ---------------------------------------------------------------------------
// Kernel 4: deterministic global reduction -> slide_logits, slide_prob.
// ---------------------------------------------------------------------------
__global__ __launch_bounds__(256) void clam_k4_reduce(const float* __restrict__ scores,
                                                      const float* __restrict__ f0,
                                                      const float* __restrict__ f1,
                                                      float* __restrict__ out, int N) {
  __shared__ float s0[256], s1[256], s2[256];
  int t = threadIdx.x;
  float a = 0.f, b = 0.f, c = 0.f;
  for (int i = t; i < N; i += 256) {                 // fixed strided order
    float sig = 1.f / (1.f + expf(-scores[i]));
    a += sig; b += f0[i]; c += f1[i];
  }
  s0[t] = a; s1[t] = b; s2[t] = c;
  __syncthreads();
  for (int s = 128; s > 0; s >>= 1) {
    if (t < s) { s0[t] += s0[t+s]; s1[t] += s1[t+s]; s2[t] += s2[t+s]; }
    __syncthreads();
  }
  if (t == 0) {
    float S = s0[0];
    float l0 = s1[0] / S, l1 = s2[0] / S;
    out[0] = l0; out[1] = l1;
    float m = fmaxf(l0, l1);
    float e0 = expf(l0 - m), e1 = expf(l1 - m), d = e0 + e1;
    out[2] = e0 / d; out[3] = e1 / d;
  }
}

// ---------------------------------------------------------------------------
// Kernel 5: top-8 / bottom-8 of scores, instance logits from bf16 feat, mean CE.
// ---------------------------------------------------------------------------
__global__ __launch_bounds__(256) void clam_k5_inst(const float* __restrict__ scores,
                                                    const unsigned short* __restrict__ feat,
                                                    const float* __restrict__ Winst,
                                                    const float* __restrict__ binst,
                                                    const int* __restrict__ labelp,
                                                    float* __restrict__ out, int N) {
  __shared__ float ltv[2048]; __shared__ int lti[2048];
  __shared__ float lbv[2048]; __shared__ int lbi[2048];
  __shared__ int   sel[16];
  __shared__ float il[32];
  int t = threadIdx.x;
  float mtv[8]; int mti[8]; float mbv[8]; int mbi[8];
#pragma unroll
  for (int j = 0; j < 8; j++) {
    mtv[j] = -__builtin_inff(); mti[j] = 0;
    mbv[j] =  __builtin_inff(); mbi[j] = 0;
  }
  for (int i = t; i < N; i += 256) {
    float s = scores[i];
    { float cv = s; int ci = i;                      // descending insert
#pragma unroll
      for (int j = 0; j < 8; j++)
        if (cv > mtv[j]) { float tv=mtv[j]; int ti=mti[j]; mtv[j]=cv; mti[j]=ci; cv=tv; ci=ti; } }
    { float cv = s; int ci = i;                      // ascending insert
#pragma unroll
      for (int j = 0; j < 8; j++)
        if (cv < mbv[j]) { float tv=mbv[j]; int ti=mbi[j]; mbv[j]=cv; mbi[j]=ci; cv=tv; ci=ti; } }
  }
#pragma unroll
  for (int j = 0; j < 8; j++) {
    ltv[t*8+j]=mtv[j]; lti[t*8+j]=mti[j]; lbv[t*8+j]=mbv[j]; lbi[t*8+j]=mbi[j];
  }
  __syncthreads();
  if (t == 0) {
    float fv[8]; int fi[8];
#pragma unroll
    for (int j = 0; j < 8; j++) { fv[j] = -__builtin_inff(); fi[j] = 0; }
    for (int e = 0; e < 2048; e++) {
      float cv = ltv[e]; int ci = lti[e];
#pragma unroll
      for (int j = 0; j < 8; j++)
        if (cv > fv[j]) { float tv=fv[j]; int ti=fi[j]; fv[j]=cv; fi[j]=ci; cv=tv; ci=ti; }
    }
#pragma unroll
    for (int j = 0; j < 8; j++) sel[j] = fi[j];
#pragma unroll
    for (int j = 0; j < 8; j++) { fv[j] = __builtin_inff(); fi[j] = 0; }
    for (int e = 0; e < 2048; e++) {
      float cv = lbv[e]; int ci = lbi[e];
#pragma unroll
      for (int j = 0; j < 8; j++)
        if (cv < fv[j]) { float tv=fv[j]; int ti=fi[j]; fv[j]=cv; fi[j]=ci; cv=tv; ci=ti; }
    }
#pragma unroll
    for (int j = 0; j < 8; j++) sel[8 + j] = fi[j];
  }
  __syncthreads();
  if (t < 32) {
    int j = t & 15, c = t >> 4;
    const unsigned short* fp = feat + (size_t)sel[j] * 512;
    float s = 0.f;
    for (int k = 0; k < 512; k++) s += bf2f(fp[k]) * Winst[k * 2 + c];
    il[j * 2 + c] = s + binst[c];
  }
  __syncthreads();
  if (t == 0) {
    float loss = 0.f;
    for (int j = 0; j < 16; j++) {
      float z0 = il[j*2+0], z1 = il[j*2+1];
      float m = fmaxf(z0, z1);
      float lse = m + logf(expf(z0 - m) + expf(z1 - m));
      float zt = (j < 8) ? z1 : z0;                  // top->class1, bottom->class0
      loss += (lse - zt);
    }
    loss *= (1.f / 16.f);
    out[4] = (labelp[0] == 1) ? loss : 0.f;
  }
}

// ---------------------------------------------------------------------------
extern "C" void kernel_launch(void* const* d_in, const int* in_sizes, int n_in,
                              void* d_out, int out_size, void* d_ws, size_t ws_size,
                              hipStream_t stream) {
  const float* h     = (const float*)d_in[0];
  const float* W1    = (const float*)d_in[1];
  const float* b1    = (const float*)d_in[2];
  const float* Wa    = (const float*)d_in[3];
  const float* ba    = (const float*)d_in[4];
  const float* Wb    = (const float*)d_in[5];
  const float* bb    = (const float*)d_in[6];
  const float* Wc    = (const float*)d_in[7];
  const float* bc    = (const float*)d_in[8];
  const float* Wcls  = (const float*)d_in[9];
  const float* bcls  = (const float*)d_in[10];
  const float* Winst = (const float*)d_in[11];
  const float* binst = (const float*)d_in[12];
  const int*   label = (const int*)d_in[13];
  float* out = (float*)d_out;

  char* ws = (char*)d_ws;
  size_t off = 0;
  auto take = [&](size_t bytes) -> char* {
    char* p = ws + off;
    off = (off + bytes + 255) & ~(size_t)255;
    return p;
  };
  unsigned short* featb = (unsigned short*)take((size_t)NROWS * 512 * 2);
  unsigned short* W1p   = (unsigned short*)take((size_t)1024 * 512 * 2);
  unsigned short* Wap   = (unsigned short*)take((size_t)512 * 256 * 2);
  unsigned short* Wbp   = (unsigned short*)take((size_t)512 * 256 * 2);
  float* scores = (float*)take((size_t)NROWS * 4);
  float* f0     = (float*)take((size_t)NROWS * 4);
  float* f1     = (float*)take((size_t)NROWS * 4);

  clam_pack_b<<<dim3((1024 * 512 + 255) / 256), dim3(256), 0, stream>>>(W1, W1p, 1024, 512);
  clam_pack_b<<<dim3((512 * 256 + 255) / 256), dim3(256), 0, stream>>>(Wa, Wap, 512, 256);
  clam_pack_b<<<dim3((512 * 256 + 255) / 256), dim3(256), 0, stream>>>(Wb, Wbp, 512, 256);

  clam_k1_feat<<<dim3(MTILES), dim3(256), 0, stream>>>(h, W1p, b1, featb);
  clam_k2_attn<<<dim3(MTILES), dim3(256), 0, stream>>>(featb, Wap, Wbp, ba, bb, Wc, bc,
                                                       Wcls, bcls, scores, f0, f1);
  clam_k4_reduce<<<dim3(1), dim3(256), 0, stream>>>(scores, f0, f1, out, NROWS);
  clam_k5_inst<<<dim3(1), dim3(256), 0, stream>>>(scores, featb, Winst, binst, label,
                                                  out, NROWS);
}
